// PointTransformerClsClean_89910845375349
// MI455X (gfx1250) — matrix operand order repaired and checked by reading, not compiled, not run on hardware
//
#include <hip/hip_runtime.h>
#include <hip/hip_bf16.h>

typedef __bf16 bf16_t;
typedef __attribute__((ext_vector_type(16))) __bf16 v16bf;
typedef __attribute__((ext_vector_type(8)))  float  v8f;
typedef __attribute__((ext_vector_type(2)))  unsigned uint2v;
typedef __attribute__((ext_vector_type(4)))  unsigned uint4v;

#define CDIV(a,b) (((a)+(b)-1)/(b))

__device__ __forceinline__ unsigned pkbf(float a, float b) {
    union { bf16_t h[2]; unsigned u; } p;
    p.h[0] = (bf16_t)a; p.h[1] = (bf16_t)b;
    return p.u;
}

// ---------------------------------------------------------------------------
// WMMA bf16 GEMM, 256 threads = 8 waves, block tile 128(M) x 64(N), BK=32,
// double-buffered LDS (one barrier per k-step) + global prefetch 2 steps
// ahead. Each wave: one 16-row slice, four accumulators (N quarters),
// 4 WMMAs / k-step sharing one A fragment.
// A-frag (16x32): lane row = lane&15, half = lane>>4; dwords:
//   q[0] = row uint4 [2*half], q[1] = row uint4 [2*half+1].
// B-frag (Bt stored [n][k]): q[0] = row uint4 [2*half], q[1] = [2*half+1].
// C/D: acc[r] -> row (r + 8*half), col (lane&15).
// ---------------------------------------------------------------------------

#define WMMA_BF16(a, b, c) \
    __builtin_amdgcn_wmma_f32_16x16x32_bf16(false, (a), false, (b), (short)0, (c), false, false)

union FragU { v16bf v; uint4v q[2]; };

__global__ void __launch_bounds__(256)
gemm_bias_kernel(const float* __restrict__ A, int lda,
                 const float* __restrict__ W,      // [K, N], N % 64 == 0
                 const float* __restrict__ bias,   // [N] or nullptr
                 float* __restrict__ Y, int ldy,
                 int M, int K, int N)               // K % 32 == 0
{
    __shared__ __align__(16) bf16_t At[2][128][32];
    __shared__ __align__(16) bf16_t Bt[2][64][32];
    const int tid  = threadIdx.x;
    const int lane = tid & 31;
    const int wave = tid >> 5;
    const int n0   = blockIdx.x * 64;
    const int m0   = blockIdx.y * 128;
    const int half = lane >> 4;
    const int l15  = lane & 15;
    const int mw   = wave * 16;
    // fill indices
    const int bk = tid >> 3;          // 0..31  (B: k index)
    const int bn = (tid & 7) * 4;     // 0..28  (B: n base, +32 on 2nd pass)
    const int ar = tid >> 3;          // 0..31  (A: row base per e-step)
    const int ac = (tid & 7) * 4;     // A: col base
    const bool full = (m0 + 128 <= M);

    float4 aR[4], bR[2];
    auto load_tiles = [&](int k0) {
        bR[0] = *(const float4*)&W[(size_t)(k0 + bk) * N + (n0 + bn)];
        bR[1] = *(const float4*)&W[(size_t)(k0 + bk) * N + (n0 + 32 + bn)];
        if (full) {
#pragma unroll
            for (int e = 0; e < 4; ++e)
                aR[e] = *(const float4*)&A[(size_t)(m0 + e * 32 + ar) * lda + (k0 + ac)];
        } else {
#pragma unroll
            for (int e = 0; e < 4; ++e) {
                float4 z = {0.f, 0.f, 0.f, 0.f};
                if (m0 + e * 32 + ar < M)
                    z = *(const float4*)&A[(size_t)(m0 + e * 32 + ar) * lda + (k0 + ac)];
                aR[e] = z;
            }
        }
    };
    auto prefetch_tiles = [&](int k0) {   // lowers to global_prefetch_b8
        __builtin_prefetch(&W[(size_t)(k0 + bk) * N + (n0 + bn)], 0, 3);
        if (full) {
#pragma unroll
            for (int e = 0; e < 4; ++e)
                __builtin_prefetch(&A[(size_t)(m0 + e * 32 + ar) * lda + (k0 + ac)], 0, 3);
        }
    };
    auto store_tiles = [&](int p) {
        Bt[p][bn + 0][bk] = (bf16_t)bR[0].x;
        Bt[p][bn + 1][bk] = (bf16_t)bR[0].y;
        Bt[p][bn + 2][bk] = (bf16_t)bR[0].z;
        Bt[p][bn + 3][bk] = (bf16_t)bR[0].w;
        Bt[p][32 + bn + 0][bk] = (bf16_t)bR[1].x;
        Bt[p][32 + bn + 1][bk] = (bf16_t)bR[1].y;
        Bt[p][32 + bn + 2][bk] = (bf16_t)bR[1].z;
        Bt[p][32 + bn + 3][bk] = (bf16_t)bR[1].w;
#pragma unroll
        for (int e = 0; e < 4; ++e) {
            uint2v pk2; pk2.x = pkbf(aR[e].x, aR[e].y); pk2.y = pkbf(aR[e].z, aR[e].w);
            *(uint2v*)&At[p][e * 32 + ar][ac] = pk2;
        }
    };

    v8f acc[4] = {{}, {}, {}, {}};
    load_tiles(0);
    store_tiles(0);
    __syncthreads();
    int p = 0;
    for (int k0 = 0; k0 < K; k0 += 32) {
        const bool has_next = (k0 + 32 < K);
        if (has_next) load_tiles(k0 + 32);
        if (k0 + 64 < K) prefetch_tiles(k0 + 64);
        FragU af, bf0, bf1, bf2, bf3;
        const uint4v* Ar4 = (const uint4v*)&At[p][mw + l15][0];
        af.q[0] = Ar4[2 * half];
        af.q[1] = Ar4[2 * half + 1];
        const uint4v* B0 = (const uint4v*)&Bt[p][l15][0];
        const uint4v* B1 = (const uint4v*)&Bt[p][16 + l15][0];
        const uint4v* B2 = (const uint4v*)&Bt[p][32 + l15][0];
        const uint4v* B3 = (const uint4v*)&Bt[p][48 + l15][0];
        bf0.q[0] = B0[2 * half]; bf0.q[1] = B0[2 * half + 1];
        bf1.q[0] = B1[2 * half]; bf1.q[1] = B1[2 * half + 1];
        bf2.q[0] = B2[2 * half]; bf2.q[1] = B2[2 * half + 1];
        bf3.q[0] = B3[2 * half]; bf3.q[1] = B3[2 * half + 1];
        acc[0] = WMMA_BF16(af.v, bf0.v, acc[0]);
        acc[1] = WMMA_BF16(af.v, bf1.v, acc[1]);
        acc[2] = WMMA_BF16(af.v, bf2.v, acc[2]);
        acc[3] = WMMA_BF16(af.v, bf3.v, acc[3]);
        if (has_next) store_tiles(p ^ 1);
        __syncthreads();
        p ^= 1;
    }
    if (full) {
#pragma unroll
        for (int nh = 0; nh < 4; ++nh) {
            const int n = n0 + nh * 16 + l15;
            const float bv = bias ? bias[n] : 0.f;
#pragma unroll
            for (int r = 0; r < 8; ++r) {
                int row = m0 + mw + r + half * 8;
                Y[(size_t)row * ldy + n] = acc[nh][r] + bv;
            }
        }
    } else {
#pragma unroll
        for (int nh = 0; nh < 4; ++nh) {
            const int n = n0 + nh * 16 + l15;
            const float bv = bias ? bias[n] : 0.f;
#pragma unroll
            for (int r = 0; r < 8; ++r) {
                int row = m0 + mw + r + half * 8;
                if (row < M) Y[(size_t)row * ldy + n] = acc[nh][r] + bv;
            }
        }
    }
}

// Batched E[b] = Q[b] @ Q[b]^T  (Q: [B,P,C], E: [B,P,P]); P%128==0, C%32==0
__global__ void __launch_bounds__(256)
gemm_nt_kernel(const float* __restrict__ Q, float* __restrict__ E, int P, int C)
{
    __shared__ __align__(16) bf16_t At[2][128][32];
    __shared__ __align__(16) bf16_t Bt[2][64][32];
    const int tid = threadIdx.x, lane = tid & 31, wave = tid >> 5;
    const float* Qb = Q + (size_t)blockIdx.z * P * C;
    float*       Eb = E + (size_t)blockIdx.z * P * P;
    const int n0 = blockIdx.x * 64;
    const int m0 = blockIdx.y * 128;
    const int half = lane >> 4, l15 = lane & 15, mw = wave * 16;
    const int bn = tid >> 3;          // B: n index (row of Q), +32 on 2nd pass
    const int bkc = (tid & 7) * 4;    // B: k base
    const int ar = tid >> 3;
    const int ac = (tid & 7) * 4;

    float4 aR[4], bR[2];
    auto load_tiles = [&](int k0) {
        bR[0] = *(const float4*)&Qb[(size_t)(n0 + bn) * C + (k0 + bkc)];
        bR[1] = *(const float4*)&Qb[(size_t)(n0 + 32 + bn) * C + (k0 + bkc)];
#pragma unroll
        for (int e = 0; e < 4; ++e)
            aR[e] = *(const float4*)&Qb[(size_t)(m0 + e * 32 + ar) * C + (k0 + ac)];
    };
    auto store_tiles = [&](int p) {
        uint2v p0; p0.x = pkbf(bR[0].x, bR[0].y); p0.y = pkbf(bR[0].z, bR[0].w);
        *(uint2v*)&Bt[p][bn][bkc] = p0;
        uint2v p1; p1.x = pkbf(bR[1].x, bR[1].y); p1.y = pkbf(bR[1].z, bR[1].w);
        *(uint2v*)&Bt[p][32 + bn][bkc] = p1;
#pragma unroll
        for (int e = 0; e < 4; ++e) {
            uint2v pk2; pk2.x = pkbf(aR[e].x, aR[e].y); pk2.y = pkbf(aR[e].z, aR[e].w);
            *(uint2v*)&At[p][e * 32 + ar][ac] = pk2;
        }
    };

    v8f acc[4] = {{}, {}, {}, {}};
    load_tiles(0);
    store_tiles(0);
    __syncthreads();
    int p = 0;
    for (int k0 = 0; k0 < C; k0 += 32) {
        const bool has_next = (k0 + 32 < C);
        if (has_next) load_tiles(k0 + 32);
        FragU af, bf0, bf1, bf2, bf3;
        const uint4v* Ar4 = (const uint4v*)&At[p][mw + l15][0];
        af.q[0] = Ar4[2 * half];
        af.q[1] = Ar4[2 * half + 1];
        const uint4v* B0 = (const uint4v*)&Bt[p][l15][0];
        const uint4v* B1 = (const uint4v*)&Bt[p][16 + l15][0];
        const uint4v* B2 = (const uint4v*)&Bt[p][32 + l15][0];
        const uint4v* B3 = (const uint4v*)&Bt[p][48 + l15][0];
        bf0.q[0] = B0[2 * half]; bf0.q[1] = B0[2 * half + 1];
        bf1.q[0] = B1[2 * half]; bf1.q[1] = B1[2 * half + 1];
        bf2.q[0] = B2[2 * half]; bf2.q[1] = B2[2 * half + 1];
        bf3.q[0] = B3[2 * half]; bf3.q[1] = B3[2 * half + 1];
        acc[0] = WMMA_BF16(af.v, bf0.v, acc[0]);
        acc[1] = WMMA_BF16(af.v, bf1.v, acc[1]);
        acc[2] = WMMA_BF16(af.v, bf2.v, acc[2]);
        acc[3] = WMMA_BF16(af.v, bf3.v, acc[3]);
        if (has_next) store_tiles(p ^ 1);
        __syncthreads();
        p ^= 1;
    }
#pragma unroll
    for (int nh = 0; nh < 4; ++nh) {
        const int n = n0 + nh * 16 + l15;
#pragma unroll
        for (int r = 0; r < 8; ++r) {
            int row = m0 + mw + r + half * 8;
            Eb[(size_t)row * P + n] = acc[nh][r];
        }
    }
}

// Batched Xr[b] = Att[b]^T @ V[b]  (Att: [B,P,P], V: [B,P,C]); P%128==0, C%64==0
__global__ void __launch_bounds__(256)
gemm_tn_kernel(const float* __restrict__ Att, const float* __restrict__ V,
               float* __restrict__ Xr, int P, int C)
{
    __shared__ __align__(16) bf16_t At[2][128][32];
    __shared__ __align__(16) bf16_t Bt[2][64][32];
    const int tid = threadIdx.x, lane = tid & 31, wave = tid >> 5;
    const float* Ab = Att + (size_t)blockIdx.z * P * P;
    const float* Vb = V   + (size_t)blockIdx.z * P * C;
    float*       Xb = Xr  + (size_t)blockIdx.z * P * C;
    const int n0 = blockIdx.x * 64;            // over C
    const int m0 = blockIdx.y * 128;           // over P (output rows)
    const int half = lane >> 4, l15 = lane & 15, mw = wave * 16;
    const int bk = tid >> 3;                   // B: k index
    const int bn = (tid & 7) * 4;              // B: n base, +32 on 2nd pass
    const int akk = tid >> 5;                  // A: k sub-index (0..7)
    const int ar4 = (tid & 31) * 4;            // A: row base

    float4 aR[4], bR[2];
    auto load_tiles = [&](int k0) {
        bR[0] = *(const float4*)&Vb[(size_t)(k0 + bk) * C + (n0 + bn)];
        bR[1] = *(const float4*)&Vb[(size_t)(k0 + bk) * C + (n0 + 32 + bn)];
#pragma unroll
        for (int e = 0; e < 4; ++e)
            aR[e] = *(const float4*)&Ab[(size_t)(k0 + e * 8 + akk) * P + (m0 + ar4)];
    };
    auto prefetch_tiles = [&](int k0) {
        __builtin_prefetch(&Vb[(size_t)(k0 + bk) * C + (n0 + bn)], 0, 3);
#pragma unroll
        for (int e = 0; e < 4; ++e)
            __builtin_prefetch(&Ab[(size_t)(k0 + e * 8 + akk) * P + (m0 + ar4)], 0, 3);
    };
    auto store_tiles = [&](int p) {
        Bt[p][bn + 0][bk] = (bf16_t)bR[0].x;
        Bt[p][bn + 1][bk] = (bf16_t)bR[0].y;
        Bt[p][bn + 2][bk] = (bf16_t)bR[0].z;
        Bt[p][bn + 3][bk] = (bf16_t)bR[0].w;
        Bt[p][32 + bn + 0][bk] = (bf16_t)bR[1].x;
        Bt[p][32 + bn + 1][bk] = (bf16_t)bR[1].y;
        Bt[p][32 + bn + 2][bk] = (bf16_t)bR[1].z;
        Bt[p][32 + bn + 3][bk] = (bf16_t)bR[1].w;
#pragma unroll
        for (int e = 0; e < 4; ++e) {
            int kk = e * 8 + akk;
            At[p][ar4 + 0][kk] = (bf16_t)aR[e].x;
            At[p][ar4 + 1][kk] = (bf16_t)aR[e].y;
            At[p][ar4 + 2][kk] = (bf16_t)aR[e].z;
            At[p][ar4 + 3][kk] = (bf16_t)aR[e].w;
        }
    };

    v8f acc[4] = {{}, {}, {}, {}};
    load_tiles(0);
    store_tiles(0);
    __syncthreads();
    int p = 0;
    for (int k0 = 0; k0 < P; k0 += 32) {
        const bool has_next = (k0 + 32 < P);
        if (has_next) load_tiles(k0 + 32);
        if (k0 + 64 < P) prefetch_tiles(k0 + 64);
        FragU af, bf0, bf1, bf2, bf3;
        const uint4v* Ar4 = (const uint4v*)&At[p][mw + l15][0];
        af.q[0] = Ar4[2 * half];
        af.q[1] = Ar4[2 * half + 1];
        const uint4v* B0 = (const uint4v*)&Bt[p][l15][0];
        const uint4v* B1 = (const uint4v*)&Bt[p][16 + l15][0];
        const uint4v* B2 = (const uint4v*)&Bt[p][32 + l15][0];
        const uint4v* B3 = (const uint4v*)&Bt[p][48 + l15][0];
        bf0.q[0] = B0[2 * half]; bf0.q[1] = B0[2 * half + 1];
        bf1.q[0] = B1[2 * half]; bf1.q[1] = B1[2 * half + 1];
        bf2.q[0] = B2[2 * half]; bf2.q[1] = B2[2 * half + 1];
        bf3.q[0] = B3[2 * half]; bf3.q[1] = B3[2 * half + 1];
        acc[0] = WMMA_BF16(af.v, bf0.v, acc[0]);
        acc[1] = WMMA_BF16(af.v, bf1.v, acc[1]);
        acc[2] = WMMA_BF16(af.v, bf2.v, acc[2]);
        acc[3] = WMMA_BF16(af.v, bf3.v, acc[3]);
        if (has_next) store_tiles(p ^ 1);
        __syncthreads();
        p ^= 1;
    }
#pragma unroll
    for (int nh = 0; nh < 4; ++nh) {
        const int n = n0 + nh * 16 + l15;
#pragma unroll
        for (int r = 0; r < 8; ++r) {
            int row = m0 + mw + r + half * 8;
            Xb[(size_t)row * C + n] = acc[nh][r];
        }
    }
}

// ---------------------------------------------------------------------------
// BatchNorm (training-mode stats) — deterministic fixed-tree reduction
// stats[c] = sum, stats[C+c] = sumsq
// ---------------------------------------------------------------------------
__global__ void bn_stats_kernel(const float* __restrict__ X, long R, int C,
                                float* __restrict__ stats)
{
    int c = blockIdx.x;
    int tid = threadIdx.x;
    __shared__ float s1[256], s2[256];
    float a = 0.f, b = 0.f;
    for (long r = tid; r < R; r += 256) {
        float x = X[r * (long)C + c];
        a += x; b += x * x;
    }
    s1[tid] = a; s2[tid] = b; __syncthreads();
    for (int s = 128; s > 0; s >>= 1) {
        if (tid < s) { s1[tid] += s1[tid + s]; s2[tid] += s2[tid + s]; }
        __syncthreads();
    }
    if (tid == 0) { stats[c] = s1[0]; stats[C + c] = s2[0]; }
}

__global__ void bn_relu_kernel(float* __restrict__ X, long total, int C,
                               const float* __restrict__ stats,
                               const float* __restrict__ g,
                               const float* __restrict__ b, float Rn)
{
    long i = (long)blockIdx.x * blockDim.x + threadIdx.x;
    if (i >= total) return;
    int c = (int)(i % C);
    float m = stats[c] / Rn;
    float v = stats[C + c] / Rn - m * m;
    float y = (X[i] - m) * rsqrtf(v + 1e-5f) * g[c] + b[c];
    X[i] = fmaxf(y, 0.f);
}

// out[bm, c] = max_k relu(bn(X[bm, k, c]))
__global__ void bn_relu_max_kernel(const float* __restrict__ X, int BM, int Kn, int C,
                                   const float* __restrict__ stats,
                                   const float* __restrict__ g,
                                   const float* __restrict__ b,
                                   float Rn, float* __restrict__ out)
{
    int i = blockIdx.x * blockDim.x + threadIdx.x;
    if (i >= BM * C) return;
    int bm = i / C, c = i % C;
    float m  = stats[c] / Rn;
    float sc = rsqrtf(stats[C + c] / Rn - m * m + 1e-5f) * g[c];
    float bb = b[c];
    float mx = 0.f;  // relu outputs are >= 0
    for (int kk = 0; kk < Kn; ++kk) {
        float x = X[((size_t)bm * Kn + kk) * C + c];
        mx = fmaxf(mx, fmaxf((x - m) * sc + bb, 0.f));
    }
    out[i] = mx;
}

// Out[r, c] = Xin[r, c] + relu(bn(T[r, c]))
__global__ void bn_relu_add_kernel(const float* __restrict__ T,
                                   const float* __restrict__ Xin, int ldx,
                                   float* __restrict__ Out, int ldo, long R, int C,
                                   const float* __restrict__ stats,
                                   const float* __restrict__ g,
                                   const float* __restrict__ b, float Rn)
{
    long i = (long)blockIdx.x * blockDim.x + threadIdx.x;
    if (i >= R * (long)C) return;
    long r = i / C; int c = (int)(i % C);
    float m = stats[c] / Rn;
    float y = (T[i] - m) * rsqrtf(stats[C + c] / Rn - m * m + 1e-5f) * g[c] + b[c];
    Out[r * (long)ldo + c] = Xin[r * (long)ldx + c] + fmaxf(y, 0.f);
}

// ---------------------------------------------------------------------------
// Elementwise / gather / reduction helpers
// ---------------------------------------------------------------------------
__global__ void gather_pos_kernel(const float* __restrict__ pts,
                                  const int* __restrict__ idx0,
                                  float* __restrict__ pos, int B, int N, int N0)
{
    int i = blockIdx.x * blockDim.x + threadIdx.x;
    if (i >= B * N0) return;
    int b = i / N0;
    int j = idx0[i];
    const float* s = pts + ((size_t)b * N + j) * 3;
    float* d = pos + (size_t)i * 3;
    d[0] = s[0]; d[1] = s[1]; d[2] = s[2];
}

__global__ void emb1_kernel(const float* __restrict__ pos,
                            const float* __restrict__ w,  // [3,64]
                            const float* __restrict__ bias,
                            float* __restrict__ Y, int R)
{
    int i = blockIdx.x * blockDim.x + threadIdx.x;
    if (i >= R * 64) return;
    int r = i / 64, c = i % 64;
    float y = bias[c];
    const float* p = pos + (size_t)r * 3;
    y += p[0] * w[c] + p[1] * w[64 + c] + p[2] * w[128 + c];
    Y[i] = y;
}

// X[b,m,kk,0:C] = feat[b, idx[b,m,kk], :]; X[b,m,kk,C:2C] = feat[b, idx[b,m,0], :]
__global__ void build_group_kernel(const float* __restrict__ feat, int Np, int C,
                                   const int* __restrict__ idx, int M, int Kn,
                                   float* __restrict__ X, int B)
{
    long total = (long)B * M * Kn * 2 * C;
    long i = (long)blockIdx.x * blockDim.x + threadIdx.x;
    if (i >= total) return;
    int c2 = (int)(i % (2 * C));
    long t = i / (2 * C);
    int kk = (int)(t % Kn); t /= Kn;
    int m  = (int)(t % M);
    int b  = (int)(t / M);
    int j  = (c2 < C) ? idx[((size_t)b * M + m) * Kn + kk]
                      : idx[((size_t)b * M + m) * Kn];
    int c  = (c2 < C) ? c2 : c2 - C;
    X[i] = feat[((size_t)b * Np + j) * C + c];
}

__global__ void softmax_kernel(float* __restrict__ E, int P, float invscale)
{
    float* e = E + (size_t)blockIdx.x * P;
    int tid = threadIdx.x;
    __shared__ float red[256];
    float mx = -1e30f;
    for (int j = tid; j < P; j += 256) mx = fmaxf(mx, e[j] * invscale);
    red[tid] = mx; __syncthreads();
    for (int s = 128; s > 0; s >>= 1) {
        if (tid < s) red[tid] = fmaxf(red[tid], red[tid + s]);
        __syncthreads();
    }
    mx = red[0]; __syncthreads();
    float sum = 0.f;
    for (int j = tid; j < P; j += 256) {
        float t = __expf(e[j] * invscale - mx);
        e[j] = t; sum += t;
    }
    red[tid] = sum; __syncthreads();
    for (int s = 128; s > 0; s >>= 1) {
        if (tid < s) red[tid] += red[tid + s];
        __syncthreads();
    }
    float inv = 1.f / red[0];
    for (int j = tid; j < P; j += 256) e[j] *= inv;
}

__global__ void sub_kernel(const float* __restrict__ X, int ldx,
                           const float* __restrict__ Y2, int ldy2,
                           float* __restrict__ D, int ldd, long R, int C)
{
    long i = (long)blockIdx.x * blockDim.x + threadIdx.x;
    if (i >= R * (long)C) return;
    long r = i / C; int c = (int)(i % C);
    D[r * (long)ldd + c] = X[r * (long)ldx + c] - Y2[r * (long)ldy2 + c];
}

__global__ void copy_kernel(const float* __restrict__ S, int lds_,
                            float* __restrict__ D, int ldd, long R, int C)
{
    long i = (long)blockIdx.x * blockDim.x + threadIdx.x;
    if (i >= R * (long)C) return;
    long r = i / C; int c = (int)(i % C);
    D[r * (long)ldd + c] = S[r * (long)lds_ + c];
}

__global__ void rowmax_kernel(const float* __restrict__ X, int B, int P, int C,
                              float* __restrict__ out)
{
    int i = blockIdx.x * blockDim.x + threadIdx.x;
    if (i >= B * C) return;
    int b = i / C, c = i % C;
    float mx = -1e30f;
    for (int m = 0; m < P; ++m)
        mx = fmaxf(mx, X[((size_t)b * P + m) * C + c]);
    out[i] = mx;
}

__global__ void cls_out_kernel(const float* __restrict__ X,   // [B,256]
                               const float* __restrict__ W,   // [256,40]
                               const float* __restrict__ bias,
                               float* __restrict__ out, int B)
{
    int i = blockIdx.x * blockDim.x + threadIdx.x;
    if (i >= B * 40) return;
    int b = i / 40, j = i % 40;
    float y = bias[j];
    const float* x = X + (size_t)b * 256;
    for (int c = 0; c < 256; ++c) y += x[c] * W[c * 40 + j];
    out[i] = y;
}

// ---------------------------------------------------------------------------
// Host orchestration
// ---------------------------------------------------------------------------
static inline void run_gemm(hipStream_t s, const float* A, int lda,
                            const float* W, const float* bias,
                            float* Y, int ldy, int M, int K, int N)
{
    dim3 g(CDIV(N, 64), CDIV(M, 128), 1);
    gemm_bias_kernel<<<g, 256, 0, s>>>(A, lda, W, bias, Y, ldy, M, K, N);
}

extern "C" void kernel_launch(void* const* d_in, const int* in_sizes, int n_in,
                              void* d_out, int out_size, void* d_ws, size_t ws_size,
                              hipStream_t stream)
{
    (void)in_sizes; (void)n_in; (void)out_size; (void)ws_size;
    enum { Bn = 16, Ntot = 4096, N0c = 2048, M1c = 1024, M2c = 512, Kc = 32, Pc = 512 };
    const long R0 = (long)Bn * N0c;            // 32768
    const long R1 = (long)Bn * M1c * Kc;       // 524288
    const long R2 = (long)Bn * M2c * Kc;       // 262144
    const long RP = (long)Bn * Pc;             // 8192

    auto F = [&](int i) { return (const float*)d_in[i]; };
    const float* all_points = F(0);
    const int* idx0 = (const int*)d_in[1];
    const int* idx1 = (const int*)d_in[2];
    const int* idx2 = (const int*)d_in[3];
    // params, JAX pytree order (dict keys sorted alphabetically, lists in order)
    const float *chw0 = F(4),  *chb0 = F(5),  *chg0 = F(6),  *chbe0 = F(7);
    const float *chw1 = F(8),  *chb1 = F(9),  *chg1 = F(10), *chbe1 = F(11);
    const float *cow  = F(12), *cob  = F(13);
    const float *c1w0 = F(14), *c1b0 = F(15), *c1g0 = F(16), *c1be0 = F(17);
    const float *c1w1 = F(18), *c1b1 = F(19), *c1g1 = F(20), *c1be1 = F(21);
    const float *c2w0 = F(22), *c2b0 = F(23), *c2g0 = F(24), *c2be0 = F(25);
    const float *c2w1 = F(26), *c2b1 = F(27), *c2g1 = F(28), *c2be1 = F(29);
    const float *ew0  = F(30), *eb0  = F(31), *eg0  = F(32), *ebe0  = F(33);
    const float *ew1  = F(34), *eb1  = F(35), *eg1  = F(36), *ebe1  = F(37);
    const float *fw   = F(38), *fb   = F(39), *fg   = F(40), *fbe   = F(41);
    struct SAP { const float *b, *bt, *bv, *g, *wq, *wt, *wv; } sa[4];
    for (int i = 0; i < 4; ++i) {
        int o = 42 + 7 * i;
        sa[i] = { F(o), F(o + 1), F(o + 2), F(o + 3), F(o + 4), F(o + 5), F(o + 6) };
    }
    const float *tw0 = F(70), *tb0 = F(71), *tg0 = F(72), *tbe0 = F(73);
    const float *tw1 = F(74), *tb1 = F(75), *tg1 = F(76), *tbe1 = F(77);

    // workspace layout
    float* base = (float*)d_ws;
    size_t off = 0;
    auto alloc = [&](size_t n) { float* p = base + off; off += (n + 63) & ~(size_t)63; return p; };
    float* pos    = alloc((size_t)Bn * N0c * 3);
    float* e1     = alloc((size_t)R0 * 64);
    float* e2     = alloc((size_t)R0 * 64);
    float* feat1  = alloc((size_t)Bn * M1c * 128);
    float* feat2  = alloc((size_t)Bn * M2c * 256);
    float* pooled = alloc((size_t)Bn * 1024);
    float* c1buf  = alloc((size_t)Bn * 512);
    float* c2buf  = alloc((size_t)Bn * 256);
    float* stats  = alloc(2048);
    float* BIG0   = alloc((size_t)R1 * 128);   // 67.1M floats
    float* BIG1   = alloc((size_t)R1 * 128);
    // stage-2 buffers alias BIG0 (conv scratch is dead by then)
    float* tA       = BIG0;
    float* tB       = tA + RP * 256;
    float* q        = tB + RP * 256;
    float* att      = q + RP * 64;
    float* vbuf     = att + (size_t)Bn * Pc * Pc;
    float* xr       = vbuf + RP * 256;
    float* dbuf     = xr + RP * 256;
    float* tbuf     = dbuf + RP * 256;
    float* h        = tbuf + RP * 256;
    float* xcat     = h + RP * 1024;
    float* fuse_out = xcat + RP * 1280;

    auto bn_stats_run = [&](const float* X, long R, int C) {
        bn_stats_kernel<<<C, 256, 0, stream>>>(X, R, C, stats);
    };
    auto bn_relu_full = [&](float* X, long R, int C, const float* g, const float* be) {
        bn_stats_run(X, R, C);
        long total = R * (long)C;
        bn_relu_kernel<<<(unsigned)CDIV(total, 256), 256, 0, stream>>>(
            X, total, C, stats, g, be, (float)R);
    };

    // 1) gather + embedding MLP [3->64->64]
    gather_pos_kernel<<<CDIV(Bn * N0c, 256), 256, 0, stream>>>(all_points, idx0, pos, Bn, Ntot, N0c);
    emb1_kernel<<<(unsigned)CDIV(R0 * 64, 256), 256, 0, stream>>>(pos, ew0, eb0, e1, (int)R0);
    bn_relu_full(e1, R0, 64, eg0, ebe0);
    run_gemm(stream, e1, 64, ew1, eb1, e2, 64, (int)R0, 64, 64);
    bn_relu_full(e2, R0, 64, eg1, ebe1);

    // 2) point conv 1: [B,M1,K,128] -> MLP[128,128] -> max_k -> [B,M1,128]
    build_group_kernel<<<(unsigned)CDIV(R1 * 128, 256), 256, 0, stream>>>(e2, N0c, 64, idx1, M1c, Kc, BIG0, Bn);
    run_gemm(stream, BIG0, 128, c1w0, c1b0, BIG1, 128, (int)R1, 128, 128);
    bn_relu_full(BIG1, R1, 128, c1g0, c1be0);
    run_gemm(stream, BIG1, 128, c1w1, c1b1, BIG0, 128, (int)R1, 128, 128);
    bn_stats_run(BIG0, R1, 128);
    bn_relu_max_kernel<<<CDIV(Bn * M1c * 128, 256), 256, 0, stream>>>(
        BIG0, Bn * M1c, Kc, 128, stats, c1g1, c1be1, (float)R1, feat1);

    // 3) point conv 2: [B,M2,K,256] -> MLP[256,256] -> max_k -> [B,M2,256]
    build_group_kernel<<<(unsigned)CDIV(R2 * 256, 256), 256, 0, stream>>>(feat1, M1c, 128, idx2, M2c, Kc, BIG0, Bn);
    run_gemm(stream, BIG0, 256, c2w0, c2b0, BIG1, 256, (int)R2, 256, 256);
    bn_relu_full(BIG1, R2, 256, c2g0, c2be0);
    run_gemm(stream, BIG1, 256, c2w1, c2b1, BIG0, 256, (int)R2, 256, 256);
    bn_stats_run(BIG0, R2, 256);
    bn_relu_max_kernel<<<CDIV(Bn * M2c * 256, 256), 256, 0, stream>>>(
        BIG0, Bn * M2c, Kc, 256, stats, c2g1, c2be1, (float)R2, feat2);

    // 4) transformer input MLP [256->256->256]
    run_gemm(stream, feat2, 256, tw0, tb0, tA, 256, (int)RP, 256, 256);
    bn_relu_full(tA, RP, 256, tg0, tbe0);
    run_gemm(stream, tA, 256, tw1, tb1, tB, 256, (int)RP, 256, 256);
    bn_relu_full(tB, RP, 256, tg1, tbe1);

    // 5) 4x self-attention; output of SA i written into h[:, i*256 : (i+1)*256]
    const float* xcur = tB;
    int ldx = 256;
    for (int i = 0; i < 4; ++i) {
        run_gemm(stream, xcur, ldx, sa[i].wq, nullptr, q, 64, (int)RP, 256, 64);
        dim3 ge(Pc / 64, Pc / 128, Bn);
        gemm_nt_kernel<<<ge, 256, 0, stream>>>(q, att, Pc, 64);
        softmax_kernel<<<Bn * Pc, 256, 0, stream>>>(att, Pc, 0.125f);  // 1/sqrt(64)
        run_gemm(stream, xcur, ldx, sa[i].wv, sa[i].bv, vbuf, 256, (int)RP, 256, 256);
        dim3 gt(256 / 64, Pc / 128, Bn);
        gemm_tn_kernel<<<gt, 256, 0, stream>>>(att, vbuf, xr, Pc, 256);
        sub_kernel<<<(unsigned)CDIV(RP * 256, 256), 256, 0, stream>>>(
            xcur, ldx, xr, 256, dbuf, 256, RP, 256);
        run_gemm(stream, dbuf, 256, sa[i].wt, sa[i].bt, tbuf, 256, (int)RP, 256, 256);
        bn_stats_run(tbuf, RP, 256);
        bn_relu_add_kernel<<<(unsigned)CDIV(RP * 256, 256), 256, 0, stream>>>(
            tbuf, xcur, ldx, h + i * 256, 1024, RP, 256, stats, sa[i].g, sa[i].b, (float)RP);
        xcur = h + i * 256;
        ldx = 1024;
    }

    // 6) concat [h | feat2] -> fuse MLP [1280->1024] -> max over points
    copy_kernel<<<(unsigned)CDIV(RP * 1024, 256), 256, 0, stream>>>(h, 1024, xcat, 1280, RP, 1024);
    copy_kernel<<<(unsigned)CDIV(RP * 256, 256), 256, 0, stream>>>(feat2, 256, xcat + 1024, 1280, RP, 256);
    run_gemm(stream, xcat, 1280, fw, fb, fuse_out, 1024, (int)RP, 1280, 1024);
    bn_relu_full(fuse_out, RP, 1024, fg, fbe);
    rowmax_kernel<<<CDIV(Bn * 1024, 256), 256, 0, stream>>>(fuse_out, Bn, Pc, 1024, pooled);

    // 7) classifier head
    run_gemm(stream, pooled, 1024, chw0, chb0, c1buf, 512, Bn, 1024, 512);
    bn_relu_full(c1buf, Bn, 512, chg0, chbe0);
    run_gemm(stream, c1buf, 512, chw1, chb1, c2buf, 256, Bn, 512, 256);
    bn_relu_full(c2buf, Bn, 256, chg1, chbe1);
    cls_out_kernel<<<CDIV(Bn * 40, 256), 256, 0, stream>>>(c2buf, cow, cob, (float*)d_out, Bn);
}
